// CodaAttention_34110630265263
// MI455X (gfx1250) — matrix operand order repaired and checked by reading, not compiled, not run on hardware
//
#include <hip/hip_runtime.h>

// ---------------- problem constants ----------------
#define Bx    2
#define Tx    2048
#define DIMx  2048
#define Hx    16
#define KVHx  4
#define HDx   128
#define TDx   64
#define Sx    (TDx + Tx)          // 2112
#define NREPx (Hx / KVHx)

// ---------------- WMMA types ----------------
typedef __attribute__((ext_vector_type(16))) __bf16 v16bf;
typedef __attribute__((ext_vector_type(8)))  float  v8f;

union FragBF {
    uint4 u[2];
    v16bf v;
};

__device__ inline v8f wmma_bf16(v16bf a, v16bf b, v8f c) {
    // D = A(16x32) * B(32x16) + C, f32 accumulate
    return __builtin_amdgcn_wmma_f32_16x16x32_bf16(
        false, a, false, b, (short)0, c, false, false);
}

__device__ inline unsigned short f2bf(float f) {
    unsigned u = __float_as_uint(f);
    unsigned r = u + 0x7FFFu + ((u >> 16) & 1u);   // RNE truncation
    return (unsigned short)(r >> 16);
}

__device__ inline float red_sum16(float v) {
    for (int o = 1; o < 16; o <<= 1) v += __shfl_xor(v, o, 32);
    return v;
}
__device__ inline float red_max16(float v) {
    for (int o = 1; o < 16; o <<= 1) v = fmaxf(v, __shfl_xor(v, o, 32));
    return v;
}
__device__ inline float red_sum32(float v) {
    for (int o = 1; o < 32; o <<= 1) v += __shfl_xor(v, o, 32);
    return v;
}

// ---------------- CDNA5 async global -> LDS copy (ASYNCcnt path) ----------------
// Per-lane 16B transfer directly into LDS, bypassing VGPRs.
// LDS operand = low 32 bits of the generic shared-memory address (ISA 10.2).
__device__ inline void coda_async_b128(const unsigned short* g, unsigned short* l) {
    unsigned lds_off = (unsigned)(size_t)l;
    unsigned long long ga = (unsigned long long)(size_t)g;
    asm volatile("global_load_async_to_lds_b128 %0, %1, off"
                 :: "v"(lds_off), "v"(ga)
                 : "memory");
}
__device__ inline void coda_wait_async0() {
    asm volatile("s_wait_asynccnt 0x0" ::: "memory");
}

// ---------------- f32 -> bf16 convert ----------------
__global__ __launch_bounds__(256) void coda_cvt_bf16(const float* __restrict__ s,
                                                     unsigned short* __restrict__ d,
                                                     int n) {
    int i = blockIdx.x * 256 + threadIdx.x;
    if (i < n) d[i] = f2bf(s[i]);
}

// ---------------- generic NT bf16 GEMM: C(MxN) = A(MxK) * B(NxK)^T ----------------
// block: 256 threads = 8 waves in 2(M) x 4(N); block tile 128(M) x 256(N);
// wave tile 64x64 = 16 WMMA per 32-wide K step. Tiles staged via
// global_load_async_to_lds_b128 (ASYNCcnt).
#define LDST 40   // padded LDS row stride (ushorts) for a 32-wide K tile
__global__ __launch_bounds__(256) void coda_gemm_nt_bf16(
    const unsigned short* __restrict__ A,
    const unsigned short* __restrict__ B,
    float* __restrict__ C,
    int M, int N, int K)
{
    __shared__ __align__(16) unsigned short As[128 * LDST];
    __shared__ __align__(16) unsigned short Bs[256 * LDST];

    const int tid  = threadIdx.x;
    const int wid  = tid >> 5;
    const int lane = tid & 31;
    const int hi16 = lane >> 4;
    const int l15  = lane & 15;

    const int m0 = blockIdx.y * 128;
    const int n0 = blockIdx.x * 256;
    const int waveM = (wid & 1) * 64;   // 2 waves along M
    const int waveN = (wid >> 1) * 64;  // 4 waves along N

    v8f acc[4][4];
    for (int mi = 0; mi < 4; ++mi)
        for (int ni = 0; ni < 4; ++ni)
            acc[mi][ni] = (v8f){0.f,0.f,0.f,0.f,0.f,0.f,0.f,0.f};

    for (int k0 = 0; k0 < K; k0 += 32) {
        // async stage: A = 512 16B-chunks (2/thread), B = 1024 chunks (4/thread)
        for (int kk = 0; kk < 2; ++kk) {
            const int chunk = tid + kk * 256;
            const int row = chunk >> 2;
            const int cq  = (chunk & 3) * 8;
            coda_async_b128(A + (size_t)(m0 + row) * K + k0 + cq,
                            As + row * LDST + cq);
        }
        for (int kk = 0; kk < 4; ++kk) {
            const int chunk = tid + kk * 256;
            const int row = chunk >> 2;
            const int cq  = (chunk & 3) * 8;
            coda_async_b128(B + (size_t)(n0 + row) * K + k0 + cq,
                            Bs + row * LDST + cq);
        }
        coda_wait_async0();
        __syncthreads();

        FragBF af[4], bf[4];
        for (int mi = 0; mi < 4; ++mi) {
            const unsigned short* src = As + (waveM + mi * 16 + l15) * LDST;
            af[mi].u[0] = *(const uint4*)(src + hi16 * 8);
            af[mi].u[1] = *(const uint4*)(src + 16 + hi16 * 8);
        }
        for (int ni = 0; ni < 4; ++ni) {
            const unsigned short* src = Bs + (waveN + ni * 16 + l15) * LDST + hi16 * 8;
            bf[ni].u[0] = *(const uint4*)(src);
            bf[ni].u[1] = *(const uint4*)(src + 16);
        }
        for (int mi = 0; mi < 4; ++mi)
            for (int ni = 0; ni < 4; ++ni)
                acc[mi][ni] = wmma_bf16(af[mi].v, bf[ni].v, acc[mi][ni]);
        __syncthreads();
    }

    for (int mi = 0; mi < 4; ++mi)
        for (int ni = 0; ni < 4; ++ni)
            for (int r = 0; r < 8; ++r)
                C[(size_t)(m0 + waveM + mi * 16 + r + 8 * hi16) * N
                  + n0 + waveN + ni * 16 + l15] = acc[mi][ni][r];
}

// ---------------- RoPE + l2norm + scale for Q and K ----------------
// one wave handles one (b, t, head-task); tasks: 0..15 = q heads, 16..19 = k heads
__global__ __launch_bounds__(256) void coda_build_qk(
    const float* __restrict__ Qf,      // (B,T,H*HD)
    const float* __restrict__ Kf,      // (B,T,KVH*HD)
    const float* __restrict__ fcos,    // (T, HD/2)
    const float* __restrict__ fsin,
    const float* __restrict__ qsc,     // (H,1,1)
    const float* __restrict__ ksc,     // (KVH,1,1)
    unsigned short* __restrict__ Qbf,  // (B,H,T,HD)
    unsigned short* __restrict__ Kall) // (B,KVH,S,HD), rows [TD, S)
{
    const int gw   = (blockIdx.x * 256 + threadIdx.x) >> 5;
    const int lane = threadIdx.x & 31;
    if (gw >= Bx * Tx * (Hx + KVHx)) return;
    const int hh = gw % (Hx + KVHx);
    const int bt = gw / (Hx + KVHx);
    const int t  = bt % Tx;
    const int b  = bt / Tx;
    const bool isq = hh < Hx;

    const float* src = isq
        ? (Qf + (size_t)(b * Tx + t) * (Hx * HDx)  + hh * HDx)
        : (Kf + (size_t)(b * Tx + t) * (KVHx * HDx) + (hh - Hx) * HDx);

    const int d0 = lane * 4;
    float v0 = src[d0 + 0], v1 = src[d0 + 1], v2 = src[d0 + 2], v3 = src[d0 + 3];

    const float c0 = fcos[t * (HDx / 2) + 2 * lane];
    const float s0 = fsin[t * (HDx / 2) + 2 * lane];
    const float c1 = fcos[t * (HDx / 2) + 2 * lane + 1];
    const float s1 = fsin[t * (HDx / 2) + 2 * lane + 1];

    float r0 = v0 * c0 - v1 * s0, i0 = v0 * s0 + v1 * c0;
    float r1 = v2 * c1 - v3 * s1, i1 = v2 * s1 + v3 * c1;

    float ss = red_sum32(r0 * r0 + i0 * i0 + r1 * r1 + i1 * i1);
    const float scl = (isq ? qsc[hh] : ksc[hh - Hx]) / fmaxf(sqrtf(ss), 1e-12f);
    r0 *= scl; i0 *= scl; r1 *= scl; i1 *= scl;

    if (isq) {
        unsigned short* dst = Qbf + ((size_t)(b * Hx + hh) * Tx + t) * HDx + d0;
        dst[0] = f2bf(r0); dst[1] = f2bf(i0); dst[2] = f2bf(r1); dst[3] = f2bf(i1);
    } else {
        unsigned short* dst = Kall + ((size_t)(b * KVHx + (hh - Hx)) * Sx + TDx + t) * HDx + d0;
        dst[0] = f2bf(r0); dst[1] = f2bf(i0); dst[2] = f2bf(r1); dst[3] = f2bf(i1);
    }
}

// ---------------- V: bias add, f32 keep, bf16 transpose ----------------
// one wave per (b, t, kvh)
__global__ __launch_bounds__(256) void coda_build_v(
    const float* __restrict__ Vraw,    // (B,T,KVH*HD)
    const float* __restrict__ vbias,   // (B,T,KVH*HD)
    float* __restrict__ Vproc,         // (B,KVH,T,HD)
    unsigned short* __restrict__ Vt)   // (B,KVH,HD,S), cols [TD, S)
{
    const int gw   = (blockIdx.x * 256 + threadIdx.x) >> 5;
    const int lane = threadIdx.x & 31;
    if (gw >= Bx * Tx * KVHx) return;
    const int kvh = gw % KVHx;
    const int bt  = gw / KVHx;
    const int t   = bt % Tx;
    const int b   = bt / Tx;

    const size_t sidx = (size_t)(b * Tx + t) * (KVHx * HDx) + kvh * HDx + lane * 4;
    for (int i = 0; i < 4; ++i) {
        const int d = lane * 4 + i;
        const float v = Vraw[sidx + i] + vbias[sidx + i];
        Vproc[((size_t)(b * KVHx + kvh) * Tx + t) * HDx + d] = v;
        Vt[((size_t)(b * KVHx + kvh) * HDx + d) * Sx + TDx + t] = f2bf(v);
    }
}

// ---------------- depth_k / depth_v prefix copies ----------------
__global__ __launch_bounds__(256) void coda_build_depth(
    const float* __restrict__ dk,      // (B,KVH,TD,HD)
    const float* __restrict__ dv,      // (B,KVH,TD,HD)
    unsigned short* __restrict__ Kall, // (B,KVH,S,HD)
    unsigned short* __restrict__ Vt)   // (B,KVH,HD,S)
{
    const int i = blockIdx.x * 256 + threadIdx.x;
    const int n = Bx * KVHx * TDx * HDx;
    if (i >= n) return;
    const int d   = i % HDx;
    const int s   = (i / HDx) % TDx;
    const int bk  = i / (HDx * TDx);   // b*KVH + kvh
    Kall[((size_t)bk * Sx + s) * HDx + d] = f2bf(dk[i]);
    Vt[((size_t)bk * HDx + d) * Sx + s]   = f2bf(dv[i]);
}

// ---------------- flash attention (WMMA QK^T and P*V) ----------------
// block = 128 threads (4 waves), each wave owns 16 query rows; grid (B*H, T/64)
__global__ __launch_bounds__(128) void coda_flash_attn(
    const unsigned short* __restrict__ Qbf,  // (B,H,T,HD)
    const unsigned short* __restrict__ Kall, // (B,KVH,S,HD)
    const unsigned short* __restrict__ Vt,   // (B,KVH,HD,S)
    float* __restrict__ Y)                   // (B,H,T,HD)
{
    __shared__ __align__(16) unsigned short pLds[4][16 * 32]; // wave-private P tiles

    const int wave = threadIdx.x >> 5;
    const int lane = threadIdx.x & 31;
    const int hi16 = lane >> 4;
    const int l15  = lane & 15;

    const int bh  = blockIdx.x;           // b*H + h
    const int b   = bh / Hx;
    const int h   = bh % Hx;
    const int kvh = h / NREPx;
    const int tw  = blockIdx.y * 64 + wave * 16;

    const unsigned short* Qb = Qbf  + ((size_t)bh * Tx + tw) * HDx;
    const unsigned short* Kb = Kall + (size_t)(b * KVHx + kvh) * Sx * HDx;
    const unsigned short* Vb = Vt   + (size_t)(b * KVHx + kvh) * HDx * Sx;

    // Q A-fragments: 4 head-dim chunks of 32
    FragBF qf[4];
    for (int kc = 0; kc < 4; ++kc) {
        const unsigned short* src = Qb + (size_t)l15 * HDx + kc * 32 + hi16 * 8;
        qf[kc].u[0] = *(const uint4*)(src);
        qf[kc].u[1] = *(const uint4*)(src + 16);
    }

    v8f o[8];
    float m[8], lsum[8];
    for (int d = 0; d < 8; ++d) o[d] = (v8f){0.f,0.f,0.f,0.f,0.f,0.f,0.f,0.f};
    for (int r = 0; r < 8; ++r) { m[r] = -3.0e38f; lsum[r] = 0.f; }

    const float scale = 0.08838834764831843f;  // 1/sqrt(HD)
    const int nchunk = (tw + 15) / 32 + 1;     // causal: keys s <= t

    for (int c = 0; c < nchunk; ++c) {
        const int sc = c * 32;

        // logits: 16x32 = two 16x16 C tiles
        v8f lg[2];
        for (int nt = 0; nt < 2; ++nt) {
            v8f a = (v8f){0.f,0.f,0.f,0.f,0.f,0.f,0.f,0.f};
            for (int kc = 0; kc < 4; ++kc) {
                FragBF kf;
                const unsigned short* src =
                    Kb + (size_t)(sc + nt * 16 + l15) * HDx + kc * 32 + hi16 * 8;
                kf.u[0] = *(const uint4*)(src);
                kf.u[1] = *(const uint4*)(src + 16);
                a = wmma_bf16(qf[kc].v, kf.v, a);
            }
            lg[nt] = a;
        }

        // online softmax update (row = r + 8*hi16, col = lane&15 within subtile)
        float alpha[8];
        unsigned short* pl = pLds[wave];
        for (int r = 0; r < 8; ++r) {
            const int trow = tw + r + 8 * hi16;
            float v0 = lg[0][r] * scale;
            float v1 = lg[1][r] * scale;
            if (sc + l15 > trow)      v0 = -1e30f;
            if (sc + 16 + l15 > trow) v1 = -1e30f;
            float mx = red_max16(fmaxf(v0, v1));
            const float mn = fmaxf(m[r], mx);
            alpha[r] = __expf(m[r] - mn);
            m[r] = mn;
            const float p0 = __expf(v0 - mn);
            const float p1 = __expf(v1 - mn);
            lsum[r] = lsum[r] * alpha[r] + red_sum16(p0 + p1);
            const int row = r + 8 * hi16;
            pl[row * 32 + l15]      = f2bf(p0);
            pl[row * 32 + 16 + l15] = f2bf(p1);
        }
        for (int d = 0; d < 8; ++d)
            for (int r = 0; r < 8; ++r)
                o[d][r] *= alpha[r];

        // P as A-fragment (wave-private LDS; in-order within wave)
        FragBF pf;
        pf.u[0] = *(const uint4*)(pl + l15 * 32 + hi16 * 8);
        pf.u[1] = *(const uint4*)(pl + l15 * 32 + 16 + hi16 * 8);

        // O += P(16x32) * V(32x16 per dim-subtile)
        for (int d = 0; d < 8; ++d) {
            FragBF vf;
            const unsigned short* src = Vb + (size_t)(d * 16 + l15) * Sx + sc + hi16 * 8;
            vf.u[0] = *(const uint4*)(src);
            vf.u[1] = *(const uint4*)(src + 16);
            o[d] = wmma_bf16(pf.v, vf.v, o[d]);
        }
    }

    float* yb = Y + ((size_t)bh * Tx + tw) * HDx;
    for (int r = 0; r < 8; ++r) {
        const float inv = 1.0f / lsum[r];
        for (int d = 0; d < 8; ++d)
            yb[(size_t)(r + 8 * hi16) * HDx + d * 16 + l15] = o[d][r] * inv;
    }
}

// ---------------- v_seq projection correction + transpose to (B,T,H*HD) bf16 ----------------
// one wave per (b, h, t)
__global__ __launch_bounds__(256) void coda_correct(
    const float* __restrict__ Y,       // (B,H,T,HD)
    const float* __restrict__ Vproc,   // (B,KVH,T,HD)
    unsigned short* __restrict__ Yc)   // (B,T,H*HD)
{
    const int gw   = (blockIdx.x * 256 + threadIdx.x) >> 5;
    const int lane = threadIdx.x & 31;
    if (gw >= Bx * Hx * Tx) return;
    const int t  = gw % Tx;
    const int bh = gw / Tx;
    const int b  = bh / Hx;
    const int h  = bh % Hx;
    const int kvh = h / NREPx;

    float y[4], v[4];
    float dot = 0.f, vn = 0.f;
    for (int i = 0; i < 4; ++i) {
        const int d = lane * 4 + i;
        y[i] = Y[((size_t)bh * Tx + t) * HDx + d];
        v[i] = Vproc[((size_t)(b * KVHx + kvh) * Tx + t) * HDx + d];
        dot += y[i] * v[i];
        vn  += v[i] * v[i];
    }
    dot = red_sum32(dot);
    vn  = red_sum32(vn);
    const float coef = dot / fmaxf(vn, 1e-8f);

    unsigned short* dst = Yc + (size_t)(b * Tx + t) * (Hx * HDx) + h * HDx + lane * 4;
    for (int i = 0; i < 4; ++i) dst[i] = f2bf(y[i] - coef * v[i]);
}

// ---------------- host launcher ----------------
extern "C" void kernel_launch(void* const* d_in, const int* in_sizes, int n_in,
                              void* d_out, int out_size, void* d_ws, size_t ws_size,
                              hipStream_t stream) {
    const float* x     = (const float*)d_in[0];
    const float* fcos  = (const float*)d_in[1];
    const float* fsin  = (const float*)d_in[2];
    const float* vbias = (const float*)d_in[3];
    const float* dk    = (const float*)d_in[4];
    const float* dv    = (const float*)d_in[5];
    const float* wq    = (const float*)d_in[6];
    const float* wk    = (const float*)d_in[7];
    const float* wv    = (const float*)d_in[8];
    const float* wo    = (const float*)d_in[9];
    const float* qsc   = (const float*)d_in[10];
    const float* ksc   = (const float*)d_in[11];
    float* out = (float*)d_out;

    const size_t MT = (size_t)Bx * Tx;           // 4096 rows
    char* ws = (char*)d_ws;
    size_t off = 0;
    auto take = [&](size_t bytes) -> char* {
        char* p = ws + off;
        off += (bytes + 255) & ~(size_t)255;
        return p;
    };

    unsigned short* xbf  = (unsigned short*)take(MT * DIMx * 2);                 // 16 MB
    unsigned short* wqbf = (unsigned short*)take((size_t)Hx * HDx * DIMx * 2);   // 8 MB
    unsigned short* wkbf = (unsigned short*)take((size_t)KVHx * HDx * DIMx * 2); // 2 MB
    unsigned short* wvbf = (unsigned short*)take((size_t)KVHx * HDx * DIMx * 2); // 2 MB
    unsigned short* wobf = (unsigned short*)take((size_t)DIMx * Hx * HDx * 2);   // 8 MB
    float* qf32 = (float*)take(MT * (size_t)(Hx * HDx) * 4);                     // 32 MB
    float* kraw = (float*)take(MT * (size_t)(KVHx * HDx) * 4);                   // 8 MB
    float* vraw = (float*)take(MT * (size_t)(KVHx * HDx) * 4);                   // 8 MB
    unsigned short* qbf  = (unsigned short*)take((size_t)Bx * Hx * Tx * HDx * 2);   // 16 MB
    unsigned short* kall = (unsigned short*)take((size_t)Bx * KVHx * Sx * HDx * 2); // 4.1 MB
    unsigned short* vt   = (unsigned short*)take((size_t)Bx * KVHx * HDx * Sx * 2); // 4.1 MB

    // stream-ordered aliases (lifetimes do not overlap):
    float* vproc = (float*)xbf;              // built after last use of xbf
    float* yattn = qf32;                     // built after last use of qf32
    unsigned short* ycorr = (unsigned short*)kraw;  // spans kraw+vraw regions

    // 1) f32 -> bf16 conversions
    {
        int n;
        n = (int)(MT * DIMx);
        coda_cvt_bf16<<<(n + 255) / 256, 256, 0, stream>>>(x, xbf, n);
        n = Hx * HDx * DIMx;
        coda_cvt_bf16<<<(n + 255) / 256, 256, 0, stream>>>(wq, wqbf, n);
        n = KVHx * HDx * DIMx;
        coda_cvt_bf16<<<(n + 255) / 256, 256, 0, stream>>>(wk, wkbf, n);
        coda_cvt_bf16<<<(n + 255) / 256, 256, 0, stream>>>(wv, wvbf, n);
        n = DIMx * Hx * HDx;
        coda_cvt_bf16<<<(n + 255) / 256, 256, 0, stream>>>(wo, wobf, n);
    }

    // 2) QKV projection GEMMs (NT, bf16 WMMA, async-LDS staging, f32 out)
    coda_gemm_nt_bf16<<<dim3((Hx * HDx) / 256, (int)(MT / 128)), 256, 0, stream>>>(
        xbf, wqbf, qf32, (int)MT, Hx * HDx, DIMx);
    coda_gemm_nt_bf16<<<dim3((KVHx * HDx) / 256, (int)(MT / 128)), 256, 0, stream>>>(
        xbf, wkbf, kraw, (int)MT, KVHx * HDx, DIMx);
    coda_gemm_nt_bf16<<<dim3((KVHx * HDx) / 256, (int)(MT / 128)), 256, 0, stream>>>(
        xbf, wvbf, vraw, (int)MT, KVHx * HDx, DIMx);

    // 3) RoPE/norm/scale, V bias + transpose, depth prefix
    {
        int waves = Bx * Tx * (Hx + KVHx);
        coda_build_qk<<<(waves * 32 + 255) / 256, 256, 0, stream>>>(
            qf32, kraw, fcos, fsin, qsc, ksc, qbf, kall);
        waves = Bx * Tx * KVHx;
        coda_build_v<<<(waves * 32 + 255) / 256, 256, 0, stream>>>(
            vraw, vbias, vproc, vt);
        int n = Bx * KVHx * TDx * HDx;
        coda_build_depth<<<(n + 255) / 256, 256, 0, stream>>>(dk, dv, kall, vt);
    }

    // 4) flash attention
    coda_flash_attn<<<dim3(Bx * Hx, Tx / 64), 128, 0, stream>>>(qbf, kall, vt, yattn);

    // 5) v_seq projection correction + transpose to (B,T,H*HD) bf16
    {
        int waves = Bx * Hx * Tx;
        coda_correct<<<(waves * 32 + 255) / 256, 256, 0, stream>>>(yattn, vproc, ycorr);
    }

    // 6) output projection -> d_out (f32)
    coda_gemm_nt_bf16<<<dim3(DIMx / 256, (int)(MT / 128)), 256, 0, stream>>>(
        ycorr, wobf, out, (int)MT, DIMx, DIMx);
}
